// AttnBlock_31327491457314
// MI455X (gfx1250) — matrix operand order repaired and checked by reading, not compiled, not run on hardware
//
#include <hip/hip_runtime.h>
#include <hip/hip_bf16.h>

#define DEVI __device__ __forceinline__

typedef __attribute__((ext_vector_type(16))) __bf16 bf16x16;
typedef __attribute__((ext_vector_type(8)))  float  f32x8;

static constexpr int C  = 512;
static constexpr int L  = 2048;
static constexpr int B  = 8;
static constexpr int BL = B * L;

DEVI f32x8 wmma_bf16(bf16x16 a, bf16x16 b, f32x8 c) {
  // D = A(16x32 bf16) * B(32x16 bf16) + C(16x16 f32)
  return __builtin_amdgcn_wmma_f32_16x16x32_bf16(
      /*neg_a=*/false, a, /*neg_b=*/false, b,
      /*c_mod=*/(short)0, c, /*reuse_a=*/false, /*reuse_b=*/false);
}

// A-matrix 16x32 bf16 fragment. Lane lo holds row M=lo; element e maps to
// K = hi*8+e (e<8) or 16+hi*8+(e-8) (e>=8).  p -> &src[row*ld + kBase].
DEVI bf16x16 frag_a(const __bf16* p, int hi) {
  bf16x16 a;
  const __bf16* p0 = p + hi * 8;
  const __bf16* p1 = p + 16 + hi * 8;
#pragma unroll
  for (int e = 0; e < 8; ++e) { a[e] = p0[e]; a[e + 8] = p1[e]; }
  return a;
}

// B-matrix 32x16 bf16 fragment. Lane lo holds column N=lo; element e maps to
// K = hi*8+e (e<8) or 16+hi*8+(e-8) (e>=8).  p -> &W[n*ld + kBase] (row-major W, W[n,k] = B^T).
DEVI bf16x16 frag_b(const __bf16* p, int hi) {
  bf16x16 b;
  const __bf16* q0 = p + hi * 8;
  const __bf16* q1 = p + 16 + hi * 8;
#pragma unroll
  for (int e = 0; e < 8; ++e) { b[e] = q0[e]; b[e + 8] = q1[e]; }
  return b;
}

// Async bulk copy global -> LDS (16 bytes per lane), tracked by ASYNCcnt.
// Inline asm: portable across ROCm-7.2 / upstream toolchains (builtin arity
// differs between them).  LDS operand = 32-bit wave-relative byte offset
// (low 32 bits of a generic shared pointer).
DEVI void async_g2l_b128(const __bf16* g, __bf16* l) {
  unsigned lds_off = (unsigned)(uintptr_t)l;
  asm volatile("global_load_async_to_lds_b128 %0, %1, off"
               :: "v"(lds_off), "v"(g) : "memory");
}
DEVI void wait_asynccnt0() {
  asm volatile("s_wait_asynccnt 0x0" ::: "memory");
}

// ---------------------------------------------------------------- weights cvt
__global__ __launch_bounds__(256) void cvt_kernel(const float* __restrict__ s,
                                                  __bf16* __restrict__ d, int n) {
  int i = blockIdx.x * 256 + threadIdx.x;
  if (i < n) d[i] = (__bf16)s[i];
}

// ---------------------------------------------------------------- group norm
__global__ __launch_bounds__(256) void gn_kernel(const float* __restrict__ x,
                                                 const float* __restrict__ gw,
                                                 const float* __restrict__ gb,
                                                 __bf16* __restrict__ h_t) {
  const int b = blockIdx.x >> 5;
  const int g = blockIdx.x & 31;
  const int cBase = g * 16;
  const int tid = threadIdx.x;
  const float* xp = x + (size_t)(b * C + cBase) * L;

  float s = 0.f, s2 = 0.f;
  for (int i = tid; i < 16 * L; i += 256) {
    float v = xp[i];
    s += v; s2 += v * v;
  }
  __shared__ float r1[256], r2[256];
  r1[tid] = s; r2[tid] = s2;
  __syncthreads();
  for (int off = 128; off > 0; off >>= 1) {
    if (tid < off) { r1[tid] += r1[tid + off]; r2[tid] += r2[tid + off]; }
    __syncthreads();
  }
  const float inv_n = 1.f / (16.f * 2048.f);
  float mean = r1[0] * inv_n;
  float var  = r2[0] * inv_n - mean * mean;
  float rstd = rsqrtf(var + 1e-6f);

  for (int i = tid; i < 16 * L; i += 256) {
    int cl = i >> 11, l = i & (L - 1);
    int c = cBase + cl;
    float v = (xp[i] - mean) * rstd * gw[c] + gb[c];
    h_t[(size_t)(b * L + l) * C + c] = (__bf16)v;
  }
}

// ---------------------------------------------------------------- QKV GEMM
// Software-pipelined: fragments for k-step kb+32 are in flight while the
// three WMMAs for kb execute.
__global__ __launch_bounds__(256) void qkv_kernel(
    const __bf16* __restrict__ h_t,
    const __bf16* __restrict__ wq, const __bf16* __restrict__ wk,
    const __bf16* __restrict__ wv,
    const float* __restrict__ bq, const float* __restrict__ bk,
    const float* __restrict__ bv,
    __bf16* __restrict__ q_t, __bf16* __restrict__ k_t,
    __bf16* __restrict__ v_c) {
  const int tid = threadIdx.x;
  const int wave = tid >> 5, lane = tid & 31, lo = lane & 15, hi = lane >> 4;
  const int mSub = wave & 1, nSub = wave >> 1;
  const int posBase = blockIdx.x * 32 + mSub * 16;
  const int nBase   = blockIdx.y * 64 + nSub * 16;

  f32x8 aq = {0,0,0,0,0,0,0,0};
  f32x8 ak = {0,0,0,0,0,0,0,0};
  f32x8 av = {0,0,0,0,0,0,0,0};

  const __bf16* aRow = h_t + (size_t)(posBase + lo) * C;
  const __bf16* qRow = wq + (size_t)(nBase + lo) * C;
  const __bf16* kRow = wk + (size_t)(nBase + lo) * C;
  const __bf16* vRow = wv + (size_t)(nBase + lo) * C;

  bf16x16 A  = frag_a(aRow, hi);
  bf16x16 Bq = frag_b(qRow, hi);
  bf16x16 Bk = frag_b(kRow, hi);
  bf16x16 Bv = frag_b(vRow, hi);
#pragma unroll 5
  for (int kb = 32; kb < C; kb += 32) {
    bf16x16 An  = frag_a(aRow + kb, hi);
    bf16x16 Bqn = frag_b(qRow + kb, hi);
    bf16x16 Bkn = frag_b(kRow + kb, hi);
    bf16x16 Bvn = frag_b(vRow + kb, hi);
    aq = wmma_bf16(A, Bq, aq);
    ak = wmma_bf16(A, Bk, ak);
    av = wmma_bf16(A, Bv, av);
    A = An; Bq = Bqn; Bk = Bkn; Bv = Bvn;
  }
  aq = wmma_bf16(A, Bq, aq);
  ak = wmma_bf16(A, Bk, ak);
  av = wmma_bf16(A, Bv, av);

  const int n = nBase + lo;
  const float scale = rsqrtf((float)C);
  const float bqv = bq[n], bkv = bk[n], bvv = bv[n];
#pragma unroll
  for (int r = 0; r < 8; ++r) {
    int pos = posBase + r + hi * 8;
    q_t[(size_t)pos * C + n] = (__bf16)((aq[r] + bqv) * scale);
    k_t[(size_t)pos * C + n] = (__bf16)(ak[r] + bkv);
    int bb = pos >> 11, l = pos & (L - 1);
    v_c[(size_t)(bb * C + n) * L + l] = (__bf16)(av[r] + bvv);
  }
}

// ---------------------------------------------------------------- attention
// One block per (b, 16-row i-tile), 8 waves.  Q tile staged in LDS via
// async-to-LDS; per 128-wide j-chunk: wave w computes one 16x16 S sub-tile
// (pipelined K frags), parallel online-softmax over 256 threads, then each
// wave accumulates a 64-channel O slice (V frags prefetched under softmax).
__global__ __launch_bounds__(256) void attn_kernel(
    const __bf16* __restrict__ q_t, const __bf16* __restrict__ k_t,
    const __bf16* __restrict__ v_c, __bf16* __restrict__ h2_t) {
  __shared__ __bf16 Qs[16 * C];      // 16 KB
  __shared__ float  Ss[16 * 128];    // 8 KB
  __shared__ __bf16 Ps[16 * 128];    // 4 KB
  __shared__ float  red[256];        // 1 KB
  __shared__ float  m_s[16], l_s[16], alpha_s[16];

  const int tid = threadIdx.x;
  const int wave = tid >> 5, lane = tid & 31, lo = lane & 15, hi = lane >> 4;
  const int iBase = blockIdx.x * 16;
  const int b = blockIdx.y;

  // --- async-stage the 16x512 Q tile into LDS (4 x b128 per thread)
#pragma unroll
  for (int it = 0; it < 4; ++it) {
    int idx = tid + 256 * it;        // b128 index; 8 bf16 each
    int r = idx >> 6, c = (idx & 63) * 8;
    async_g2l_b128(&q_t[(size_t)(b * L + iBase + r) * C + c], &Qs[idx * 8]);
  }
  if (tid < 16) { m_s[tid] = -1e30f; l_s[tid] = 0.f; }

  f32x8 acc[4];
#pragma unroll
  for (int t = 0; t < 4; ++t) acc[t] = (f32x8){0,0,0,0,0,0,0,0};

  const __bf16* vBase[4];
#pragma unroll
  for (int t = 0; t < 4; ++t)
    vBase[t] = v_c + (size_t)(b * C + wave * 64 + t * 16 + lo) * L;

  wait_asynccnt0();
  __syncthreads();

  const int row = tid >> 4, seg = tid & 15;   // softmax mapping: 16 thr/row

  for (int jc = 0; jc < L; jc += 128) {
    const __bf16* kRow = k_t + (size_t)(b * L + jc + wave * 16 + lo) * C;
    if (jc + 128 < L)  // prefetch next chunk's K row into L2/L0
      __builtin_prefetch(kRow + (size_t)128 * C, 0, 0);

    // ---- S = Q^T K for this wave's 16-column j sub-tile (pipelined)
    f32x8 s = {0,0,0,0,0,0,0,0};
    bf16x16 Af = frag_a(&Qs[lo * C], hi);
    bf16x16 Bf = frag_b(kRow, hi);
#pragma unroll 5
    for (int kb = 32; kb < C; kb += 32) {
      bf16x16 An = frag_a(&Qs[lo * C + kb], hi);
      bf16x16 Bn = frag_b(kRow + kb, hi);
      s = wmma_bf16(Af, Bf, s);
      Af = An; Bf = Bn;
    }
    s = wmma_bf16(Af, Bf, s);
#pragma unroll
    for (int r = 0; r < 8; ++r)
      Ss[(r + hi * 8) * 128 + wave * 16 + lo] = s[r];

    // issue V fragment loads for kb=0 now: latency hides under softmax
    bf16x16 Vf[4];
#pragma unroll
    for (int t = 0; t < 4; ++t) Vf[t] = frag_b(vBase[t] + jc, hi);

    __syncthreads();

    // ---- parallel online softmax (16 threads per row, 8 elems each)
    {
      const float* srow = &Ss[row * 128 + seg * 8];
      float lmax = -1e30f;
#pragma unroll
      for (int e = 0; e < 8; ++e) lmax = fmaxf(lmax, srow[e]);
      red[tid] = lmax;
      __syncthreads();
      if (tid < 16) {
        float mc = m_s[tid];
        for (int u = 0; u < 16; ++u) mc = fmaxf(mc, red[tid * 16 + u]);
        alpha_s[tid] = __expf(m_s[tid] - mc);
        m_s[tid] = mc;
      }
      __syncthreads();
      float mc = m_s[row];
      float lsum = 0.f;
#pragma unroll
      for (int e = 0; e < 8; ++e) {
        float p = __expf(srow[e] - mc);
        Ps[row * 128 + seg * 8 + e] = (__bf16)p;
        lsum += p;
      }
      red[tid] = lsum;
      __syncthreads();
      if (tid < 16) {
        float ls = l_s[tid] * alpha_s[tid];
        for (int u = 0; u < 16; ++u) ls += red[tid * 16 + u];
        l_s[tid] = ls;
      }
    }

    // ---- rescale accumulators, then O += P * V (pipelined V frags)
    float al[8];
#pragma unroll
    for (int r = 0; r < 8; ++r) al[r] = alpha_s[r + hi * 8];
#pragma unroll
    for (int t = 0; t < 4; ++t)
#pragma unroll
      for (int r = 0; r < 8; ++r) acc[t][r] *= al[r];

#pragma unroll
    for (int kb = 0; kb < 128; kb += 32) {
      bf16x16 Vn[4];
      if (kb < 96) {
#pragma unroll
        for (int t = 0; t < 4; ++t)
          Vn[t] = frag_b(vBase[t] + jc + kb + 32, hi);
      }
      bf16x16 A = frag_a(&Ps[lo * 128 + kb], hi);
#pragma unroll
      for (int t = 0; t < 4; ++t) acc[t] = wmma_bf16(A, Vf[t], acc[t]);
      if (kb < 96) {
#pragma unroll
        for (int t = 0; t < 4; ++t) Vf[t] = Vn[t];
      }
    }
    // Ss/Ps/alpha hazards vs next chunk covered by the barriers above.
  }
  __syncthreads();  // final l_s update visible to all waves

  float linv[8];
#pragma unroll
  for (int r = 0; r < 8; ++r) linv[r] = 1.f / l_s[r + hi * 8];
#pragma unroll
  for (int t = 0; t < 4; ++t)
#pragma unroll
    for (int r = 0; r < 8; ++r)
      h2_t[(size_t)(b * L + iBase + r + hi * 8) * C + wave * 64 + t * 16 + lo] =
          (__bf16)(acc[t][r] * linv[r]);
}

// --------------------------------------------------- out projection + residual
__global__ __launch_bounds__(256) void proj_kernel(
    const __bf16* __restrict__ h2_t, const __bf16* __restrict__ wo,
    const float* __restrict__ bo, const float* __restrict__ x,
    float* __restrict__ out) {
  const int tid = threadIdx.x;
  const int wave = tid >> 5, lane = tid & 31, lo = lane & 15, hi = lane >> 4;
  const int mSub = wave & 1, nSub = wave >> 1;
  const int posBase = blockIdx.x * 32 + mSub * 16;
  const int nBase   = blockIdx.y * 64 + nSub * 16;

  f32x8 acc = {0,0,0,0,0,0,0,0};
  const __bf16* aRow = h2_t + (size_t)(posBase + lo) * C;
  const __bf16* wRow = wo + (size_t)(nBase + lo) * C;

  bf16x16 A = frag_a(aRow, hi);
  bf16x16 Bf = frag_b(wRow, hi);
#pragma unroll 5
  for (int kb = 32; kb < C; kb += 32) {
    bf16x16 An = frag_a(aRow + kb, hi);
    bf16x16 Bn = frag_b(wRow + kb, hi);
    acc = wmma_bf16(A, Bf, acc);
    A = An; Bf = Bn;
  }
  acc = wmma_bf16(A, Bf, acc);

  const int n = nBase + lo;
  const float bov = bo[n];
#pragma unroll
  for (int r = 0; r < 8; ++r) {
    int pos = posBase + r + hi * 8;
    int bb = pos >> 11, l = pos & (L - 1);
    size_t idx = (size_t)(bb * C + n) * L + l;
    out[idx] = x[idx] + acc[r] + bov;
  }
}

// ---------------------------------------------------------------- launcher
extern "C" void kernel_launch(void* const* d_in, const int* in_sizes, int n_in,
                              void* d_out, int out_size, void* d_ws, size_t ws_size,
                              hipStream_t stream) {
  const float* x    = (const float*)d_in[0];
  const float* gn_w = (const float*)d_in[1];
  const float* gn_b = (const float*)d_in[2];
  const float* wq   = (const float*)d_in[3];
  const float* bq   = (const float*)d_in[4];
  const float* wk   = (const float*)d_in[5];
  const float* bk   = (const float*)d_in[6];
  const float* wv   = (const float*)d_in[7];
  const float* bv   = (const float*)d_in[8];
  const float* wo   = (const float*)d_in[9];
  const float* bo   = (const float*)d_in[10];
  float* out = (float*)d_out;

  char* ws = (char*)d_ws;
  const size_t MAT = (size_t)BL * C * sizeof(__bf16);  // 16 MB
  __bf16* h_t  = (__bf16*)(ws);             // reused as h2_t after QKV
  __bf16* q_t  = (__bf16*)(ws + MAT);
  __bf16* k_t  = (__bf16*)(ws + 2 * MAT);
  __bf16* v_c  = (__bf16*)(ws + 3 * MAT);
  __bf16* wq_b = (__bf16*)(ws + 4 * MAT);
  __bf16* wk_b = wq_b + (size_t)C * C;
  __bf16* wv_b = wk_b + (size_t)C * C;
  __bf16* wo_b = wv_b + (size_t)C * C;

  const int NW = C * C;
  cvt_kernel<<<(NW + 255) / 256, 256, 0, stream>>>(wq, wq_b, NW);
  cvt_kernel<<<(NW + 255) / 256, 256, 0, stream>>>(wk, wk_b, NW);
  cvt_kernel<<<(NW + 255) / 256, 256, 0, stream>>>(wv, wv_b, NW);
  cvt_kernel<<<(NW + 255) / 256, 256, 0, stream>>>(wo, wo_b, NW);

  gn_kernel<<<B * 32, 256, 0, stream>>>(x, gn_w, gn_b, h_t);

  qkv_kernel<<<dim3(BL / 32, C / 64), 256, 0, stream>>>(
      h_t, wq_b, wk_b, wv_b, bq, bk, bv, q_t, k_t, v_c);

  __bf16* h2_t = h_t;  // h_t no longer needed
  attn_kernel<<<dim3(L / 16, B), 256, 0, stream>>>(q_t, k_t, v_c, h2_t);

  proj_kernel<<<dim3(BL / 32, C / 64), 256, 0, stream>>>(h2_t, wo_b, bo, x, out);
}
